// GATLayer_6743098654873
// MI455X (gfx1250) — compile-verified, matrix operand order checked by reading, not run on hardware
//
#include <hip/hip_runtime.h>

#define NN 8192
#define FF 256
#define BK 64
#define LALPHA 0.2f
#define NEGINF -9.0e15f

typedef __attribute__((ext_vector_type(16))) __bf16 v16bf;
typedef __attribute__((ext_vector_type(8)))  __bf16 v8bf;
typedef __attribute__((ext_vector_type(4)))  __bf16 v4bf;
typedef __attribute__((ext_vector_type(8)))  float  v8f;
typedef __attribute__((ext_vector_type(4)))  float  v4f;

static __device__ __forceinline__ v16bf cat8(v8bf lo, v8bf hi) {
    v16bf r;
#pragma unroll
    for (int e = 0; e < 8; ++e) { r[e] = lo[e]; r[8 + e] = hi[e]; }
    return r;
}

// ---------------------------------------------------------------------------
// Kernel 1: h = X @ W  (bf16 WMMA, f32 accumulate). Writes h_f32 and h_bf16.
// grid (NN/16, 4), block 128 (4 waves, one 16x16 C tile each).
// ---------------------------------------------------------------------------
__global__ void __launch_bounds__(128) gemm_h_kernel(
    const float* __restrict__ X, const float* __restrict__ W,
    float* __restrict__ hf, __bf16* __restrict__ hb) {
  const int lane = threadIdx.x & 31;
  const int wid  = threadIdx.x >> 5;
  const int row0 = blockIdx.x * 16;
  const int n0   = blockIdx.y * 64 + wid * 16;
  const int half = lane >> 4;      // 0: lanes 0-15, 1: lanes 16-31
  const int kb   = half * 8;       // K sub-base per ISA 16-bit A/B striping
  const int mA   = lane & 15;
  const int nB   = n0 + (lane & 15);

  v8f acc = {};
  for (int k0 = 0; k0 < FF; k0 += 32) {
    // A fragment: 16x32 bf16, row mA of X, K = kb..kb+7 and 16+kb..16+kb+7
    const float* xp = X + (size_t)(row0 + mA) * FF + k0;
    v4f a0 = *(const v4f*)(xp + kb);
    v4f a1 = *(const v4f*)(xp + kb + 4);
    v4f a2 = *(const v4f*)(xp + 16 + kb);
    v4f a3 = *(const v4f*)(xp + 16 + kb + 4);
    v16bf af;
#pragma unroll
    for (int e = 0; e < 4; ++e) {
      af[e]      = (__bf16)a0[e];
      af[4 + e]  = (__bf16)a1[e];
      af[8 + e]  = (__bf16)a2[e];
      af[12 + e] = (__bf16)a3[e];
    }
    // B fragment: 32x16 bf16, column nB of W (K-major gather; W is L2-resident)
    v16bf bf_;
#pragma unroll
    for (int e = 0; e < 16; ++e) {
      const int k = (e < 8) ? (kb + e) : (16 + kb + (e - 8));
      bf_[e] = (__bf16)W[(size_t)(k0 + k) * FF + nB];
    }
    acc = __builtin_amdgcn_wmma_f32_16x16x32_bf16(
        false, af, false, bf_, (short)0, acc, false, false);
  }
#pragma unroll
  for (int r = 0; r < 8; ++r) {
    const int m = r + 8 * half;                 // C layout: VGPR r -> M=r / M=8+r
    const size_t o = (size_t)(row0 + m) * FF + nB;
    hf[o] = acc[r];
    hb[o] = (__bf16)acc[r];
  }
}

// ---------------------------------------------------------------------------
// Kernel 2: f_src = h @ a[:F], f_dst = h @ a[F:]. One wave per row.
// grid NN/8, block 256 (8 waves).
// ---------------------------------------------------------------------------
__global__ void __launch_bounds__(256) fvec_kernel(
    const float* __restrict__ hf, const float* __restrict__ a,
    float* __restrict__ fs, float* __restrict__ fd) {
  const int row  = blockIdx.x * 8 + (threadIdx.x >> 5);
  const int lane = threadIdx.x & 31;
  const float* hp = hf + (size_t)row * FF + lane * 8;
  v4f h0 = *(const v4f*)(hp);
  v4f h1 = *(const v4f*)(hp + 4);
  v4f s0 = *(const v4f*)(a + lane * 8);
  v4f s1 = *(const v4f*)(a + lane * 8 + 4);
  v4f d0 = *(const v4f*)(a + FF + lane * 8);
  v4f d1 = *(const v4f*)(a + FF + lane * 8 + 4);
  float ss = 0.f, sd = 0.f;
#pragma unroll
  for (int e = 0; e < 4; ++e) {
    ss += h0[e] * s0[e] + h1[e] * s1[e];
    sd += h0[e] * d0[e] + h1[e] * d1[e];
  }
#pragma unroll
  for (int off = 16; off > 0; off >>= 1) {
    ss += __shfl_down(ss, off, 32);
    sd += __shfl_down(sd, off, 32);
  }
  if (lane == 0) { fs[row] = ss; fd[row] = sd; }
}

// ---------------------------------------------------------------------------
// Kernel 3: fused masked softmax + attention @ h (flash-attention style).
// 16 output rows per block; 4 waves x 64 cols; online softmax over 64-j blocks.
// Scores live in registers; exps spread over all 128 threads; h block staged
// transposed in LDS with packed b64 stores; P written as one b128 per thread.
// grid NN/16, block 128.
// ---------------------------------------------------------------------------
__global__ void __launch_bounds__(128) gat_attn_kernel(
    const float* __restrict__ fs, const float* __restrict__ fd,
    const int* __restrict__ adj, const __bf16* __restrict__ hb,
    float* __restrict__ out) {
  __shared__ __align__(16) __bf16 ht[FF * BK];     // transposed h: [c*BK + j]  32KB
  __shared__ __align__(16) __bf16 pbuf[16 * BK];   // bf16 P: [m*BK + k]         2KB
  __shared__ __align__(16) float  pmax[16 * 8];    // per-row partial maxima
  __shared__ __align__(16) float  psums[16 * 8];   // per-row partial exp sums
  __shared__ __align__(16) float  mrow[16];
  __shared__ __align__(16) float  lrow[16];
  __shared__ __align__(16) float  srow[16];        // per-row rescale factor
  __shared__ __align__(16) float  mnb[16];         // broadcast new max
  __shared__ __align__(16) float  fsrow[16];

  const int tid  = threadIdx.x;
  const int lane = tid & 31;
  const int wid  = tid >> 5;
  const int row0 = blockIdx.x * 16;
  const int half = lane >> 4;
  const int kb   = half * 8;

  if (tid < 16) {
    mrow[tid]  = NEGINF;   // finite, matching reference: all-masked rows -> uniform
    lrow[tid]  = 0.f;
    fsrow[tid] = fs[row0 + tid];
  }
  v8f acc[4] = {};
  __syncthreads();

  const int    sm = tid >> 3;          // score row 0..15
  const int    k8 = (tid & 7) * 8;     // score col chunk 0,8,...,56
  const size_t arowbase = (size_t)(row0 + sm) * NN;

  for (int j0 = 0; j0 < NN; j0 += BK) {
    // --- Phase 1a: masked leaky-relu scores (kept in registers) + local max ---
    float sreg[8];
    {
      const int4 a0 = *(const int4*)(adj + arowbase + j0 + k8);
      const int4 a1 = *(const int4*)(adj + arowbase + j0 + k8 + 4);
      const v4f  f0 = *(const v4f*)(fd + j0 + k8);
      const v4f  f1 = *(const v4f*)(fd + j0 + k8 + 4);
      const float fsv = fsrow[sm];
      const int   am[8] = {a0.x, a0.y, a0.z, a0.w, a1.x, a1.y, a1.z, a1.w};
      const float fv[8] = {f0[0], f0[1], f0[2], f0[3], f1[0], f1[1], f1[2], f1[3]};
      float lmax = NEGINF;
#pragma unroll
      for (int e = 0; e < 8; ++e) {
        const float x  = fsv + fv[e];
        const float lr = (x > 0.f) ? x : LALPHA * x;
        const float s  = (am[e] > 0) ? lr : NEGINF;
        sreg[e] = s;
        lmax = fmaxf(lmax, s);
      }
      pmax[sm * 8 + (tid & 7)] = lmax;
      if (j0 + BK < NN)
        __builtin_prefetch(adj + arowbase + j0 + BK + k8, 0, 1);
    }
    // --- Phase 1b: stage 64x256 h block transposed, 4 rows packed per b64 ---
#pragma unroll
    for (int u = 0; u < 4; ++u) {
      const int uu = tid + u * 128;        // 0..511 units of 8c x 4j
      const int j4 = (uu & 15) * 4;        // 0..60
      const int c8 = (uu >> 4) * 8;        // 0..248
      const __bf16* src = hb + (size_t)(j0 + j4) * FF + c8;
      v8bf r0 = *(const v8bf*)(src);
      v8bf r1 = *(const v8bf*)(src + FF);
      v8bf r2 = *(const v8bf*)(src + 2 * FF);
      v8bf r3 = *(const v8bf*)(src + 3 * FF);
#pragma unroll
      for (int e = 0; e < 8; ++e) {
        v4bf pk = {r0[e], r1[e], r2[e], r3[e]};
        *(v4bf*)(ht + (c8 + e) * BK + j4) = pk;
      }
    }
    __syncthreads();

    // --- Phase 2: per-row max reduce + rescale factor (16 threads) ---
    if (tid < 16) {
      float bmax = pmax[tid * 8];
#pragma unroll
      for (int i = 1; i < 8; ++i) bmax = fmaxf(bmax, pmax[tid * 8 + i]);
      const float mold = mrow[tid];
      const float mnew = fmaxf(mold, bmax);
      srow[tid] = __expf(mold - mnew);
      mrow[tid] = mnew;
      mnb[tid]  = mnew;
    }
    __syncthreads();

    // --- Phase 3: exps spread across all 128 threads; P as one b128 store ---
    {
      const float mn = mnb[sm];
      v8bf pk;
      float ps = 0.f;
#pragma unroll
      for (int e = 0; e < 8; ++e) {
        const float p = __expf(sreg[e] - mn);
        pk[e] = (__bf16)p;
        ps += p;
      }
      *(v8bf*)(pbuf + sm * BK + k8) = pk;
      psums[sm * 8 + (tid & 7)] = ps;
    }
    __syncthreads();

    // --- Phase 4: l update (16 threads, no extra barrier) + rescale + WMMA ---
    if (tid < 16) {
      float s = 0.f;
#pragma unroll
      for (int i = 0; i < 8; ++i) s += psums[tid * 8 + i];
      lrow[tid] = lrow[tid] * srow[tid] + s;
    }
    v16bf afr[2];
#pragma unroll
    for (int ks = 0; ks < 2; ++ks) {
      const int base = (lane & 15) * BK + ks * 32 + kb;
      v8bf lo = *(const v8bf*)(pbuf + base);
      v8bf hi = *(const v8bf*)(pbuf + base + 16);
      afr[ks] = cat8(lo, hi);
    }
    const v4f sc0 = *(const v4f*)(srow + half * 8);
    const v4f sc1 = *(const v4f*)(srow + half * 8 + 4);
#pragma unroll
    for (int tt = 0; tt < 4; ++tt) {
#pragma unroll
      for (int r = 0; r < 8; ++r) acc[tt][r] *= (r < 4) ? sc0[r] : sc1[r - 4];
      const int n = wid * 64 + tt * 16 + (lane & 15);
#pragma unroll
      for (int ks = 0; ks < 2; ++ks) {
        const int nb = n * BK + ks * 32 + kb;
        v8bf lo = *(const v8bf*)(ht + nb);
        v8bf hi = *(const v8bf*)(ht + nb + 16);
        v16bf bfr = cat8(lo, hi);
        acc[tt] = __builtin_amdgcn_wmma_f32_16x16x32_bf16(
            false, afr[ks], false, bfr, (short)0, acc[tt], false, false);
      }
    }
    __syncthreads();
  }

  // --- epilogue: normalize by l, ELU, store f32 ---
  const v4f l0 = *(const v4f*)(lrow + half * 8);
  const v4f l1 = *(const v4f*)(lrow + half * 8 + 4);
#pragma unroll
  for (int tt = 0; tt < 4; ++tt) {
    const int n = wid * 64 + tt * 16 + (lane & 15);
#pragma unroll
    for (int r = 0; r < 8; ++r) {
      const int m = r + 8 * half;
      const float l = (r < 4) ? l0[r] : l1[r - 4];
      const float x = acc[tt][r] / l;
      out[(size_t)(row0 + m) * FF + n] = (x > 0.f) ? x : (__expf(x) - 1.f);
    }
  }
}

// ---------------------------------------------------------------------------
extern "C" void kernel_launch(void* const* d_in, const int* in_sizes, int n_in,
                              void* d_out, int out_size, void* d_ws, size_t ws_size,
                              hipStream_t stream) {
  (void)in_sizes; (void)n_in; (void)out_size; (void)ws_size;
  const float* X   = (const float*)d_in[0];   // mole_out [8192,256]
  const int*   adj = (const int*)d_in[1];     // adj [8192,8192]
  const float* W   = (const float*)d_in[2];   // W [256,256]
  const float* a   = (const float*)d_in[3];   // a [512,1]

  char* ws = (char*)d_ws;
  float*  hf  = (float*)ws;                                     // 8 MB
  __bf16* hbm = (__bf16*)(ws + (size_t)NN * FF * 4);            // 4 MB
  float*  fsv = (float*)(ws + (size_t)NN * FF * 6);             // 32 KB
  float*  fdv = fsv + NN;                                       // 32 KB

  dim3 g1(NN / 16, 4);
  gemm_h_kernel<<<g1, dim3(128), 0, stream>>>(X, W, hf, hbm);
  fvec_kernel<<<dim3(NN / 8), dim3(256), 0, stream>>>(hf, a, fsv, fdv);
  gat_attn_kernel<<<dim3(NN / 16), dim3(128), 0, stream>>>(
      fsv, fdv, adj, hbm, (float*)d_out);
}
